// RNN_90881507983486
// MI455X (gfx1250) — compile-verified, matrix operand order checked by reading, not run on hardware
//
#include <hip/hip_runtime.h>
#include <hip/hip_bf16.h>

// LSTM: B=64, T=512, D=H=1024, W:(2048,4096), gates (i,j,f,o), forget bias 1.0
// Strategy:
//   1) pack_w: fp32 W -> bf16, pre-swizzled into exact WMMA B-fragment layout,
//      tile-contiguous (1KB per 32x16 K x N tile) in workspace (16 MB, L2-resident).
//   2) 512 sequential lstm_step kernels (graph-friendly). Each fuses the
//      x-part and h-part GEMM (K=2048) using v_wmma_f32_16x16x32_bf16 with f32
//      accumulate, then the gate nonlinearities + c/h update, entirely on-chip.

typedef __attribute__((ext_vector_type(16))) __bf16 v16bf;
typedef __attribute__((ext_vector_type(8)))  float  v8f;

#define BATCH 64
#define TLEN  512
#define DIN   1024
#define HDIM  1024
#define KTOT  2048      // D + H
#define NTOT  4096      // 4 * H
#define WK    32        // WMMA K per instruction (bf16)
#define NKT   (KTOT / WK)   // 64 K-tiles
#define CHUNK_KT 4          // K-tiles staged in LDS per chunk (K=128)
#define NCHUNK (NKT / CHUNK_KT)

// 16-bit operand fragment mapping (ISA 7.12.2): lane L, element e -> K
__device__ __forceinline__ int kmap(int lane, int e) {
    return ((e < 8) ? e : e + 8) + ((lane < 16) ? 0 : 8);
}

__device__ __forceinline__ unsigned short f32_to_bf16(float f) {
    union { float f; unsigned int u; } v; v.f = f;
    unsigned int u = v.u;
    u += 0x7fffu + ((u >> 16) & 1u);   // round-to-nearest-even
    return (unsigned short)(u >> 16);
}

__device__ __forceinline__ float sigmoidf(float x) {
    return 1.0f / (1.0f + __expf(-x));
}

// ---------------------------------------------------------------------------
// Pack W (fp32 row-major [2048, 4096]) into bf16 WMMA B-fragments.
// Packed element index: ((nt*64 + kt)*32 + lane)*16 + e  (nt: 16-wide N tile,
// kt: 32-deep K tile).  A lane's 16 bf16 (32 bytes) are contiguous.
// ---------------------------------------------------------------------------
__global__ __launch_bounds__(256) void pack_w(const float* __restrict__ W,
                                              unsigned short* __restrict__ Wpk) {
    int gid  = blockIdx.x * 256 + threadIdx.x;     // 0 .. 8388607
    int e    = gid & 15;
    int lane = (gid >> 4) & 31;
    int kt   = (gid >> 9) & 63;
    int nt   = gid >> 15;
    int k    = kt * WK + kmap(lane, e);
    int n    = nt * 16 + (lane & 15);
    Wpk[gid] = f32_to_bf16(W[(size_t)k * NTOT + n]);
}

// Zero h0 (slot 0) and c.
__global__ __launch_bounds__(256) void init_state(float* __restrict__ h0,
                                                  float* __restrict__ c) {
    int gid = blockIdx.x * 256 + threadIdx.x;      // 0 .. 65535
    h0[gid] = 0.0f;
    c[gid]  = 0.0f;
}

// ---------------------------------------------------------------------------
// One LSTM timestep. Grid: 64 blocks x 256 threads (8 waves).
// Block bx owns h-columns [bx*16, bx*16+16) across all 4 gates and all 64 rows.
// Wave w: M-tile = w%4; gates {0,1} (w<4) or {2,3} (w>=4), one 16-wide N tile
// per gate. K loop = 64 WMMA K-tiles over [x_t ; h_prev].
// ---------------------------------------------------------------------------
__global__ __launch_bounds__(256) void lstm_step(
    const float* __restrict__ x,              // [B, T, D]
    const unsigned short* __restrict__ Wpk,   // packed bf16 W
    const float* __restrict__ bias,           // [4096]
    const float* __restrict__ h_prev,         // [64, 1024]
    float* __restrict__ h_next,               // [64, 1024]
    float* __restrict__ c,                    // [64, 1024] in/out
    float* __restrict__ out,                  // [B, T, H]
    int t)
{
    __shared__ __align__(32) unsigned short a_lds[CHUNK_KT * 4 * 32 * 16]; // 16 KB
    __shared__ float z_lds[64 * 64];                                       // 16 KB

    const int tid   = threadIdx.x;
    const int wave  = tid >> 5;
    const int lane  = tid & 31;
    const int bx    = blockIdx.x;           // 0..63
    const int mt    = wave & 3;             // M tile (16 rows)
    const int g0    = (wave >> 2) * 2;      // first gate handled by this wave

    const int nt0 = (g0 + 0) * 64 + bx;     // global 16-wide N tiles
    const int nt1 = (g0 + 1) * 64 + bx;
    const unsigned short* bp0 = Wpk + ((size_t)nt0 * NKT) * 512 + lane * 16;
    const unsigned short* bp1 = Wpk + ((size_t)nt1 * NKT) * 512 + lane * 16;

    v8f acc0 = {};
    v8f acc1 = {};

    for (int ch = 0; ch < NCHUNK; ++ch) {
        // ---- stage A chunk (64 rows x 128 K) as bf16 fragments in LDS ----
        // layout: ((ktl*4 + mt)*32 + lane)*16 + e
        for (int s = 0; s < 32; ++s) {
            int idx = s * 256 + tid;              // 0 .. 8191
            int e   = idx & 15;
            int ln  = (idx >> 4) & 31;
            int mtl = (idx >> 9) & 3;
            int ktl = idx >> 11;
            int m   = mtl * 16 + (ln & 15);
            int k   = ch * (CHUNK_KT * WK) + ktl * WK + kmap(ln, e);
            float v = (k < DIN) ? x[((size_t)m * TLEN + t) * DIN + k]
                                : h_prev[m * HDIM + (k - DIN)];
            a_lds[idx] = f32_to_bf16(v);
        }
        __syncthreads();

        #pragma unroll
        for (int ktl = 0; ktl < CHUNK_KT; ++ktl) {
            const int kt = ch * CHUNK_KT + ktl;
            const v16bf a  = *(const v16bf*)&a_lds[((ktl * 4 + mt) * 32 + lane) * 16];
            const v16bf b0 = *(const v16bf*)(bp0 + (size_t)kt * 512);
            const v16bf b1 = *(const v16bf*)(bp1 + (size_t)kt * 512);
            if (kt + CHUNK_KT < NKT) {   // pull next K-slab of W through L2
                __builtin_prefetch(bp0 + (size_t)(kt + CHUNK_KT) * 512, 0, 1);
                __builtin_prefetch(bp1 + (size_t)(kt + CHUNK_KT) * 512, 0, 1);
            }
            acc0 = __builtin_amdgcn_wmma_f32_16x16x32_bf16(
                       false, a, false, b0, (short)0, acc0, false, false);
            acc1 = __builtin_amdgcn_wmma_f32_16x16x32_bf16(
                       false, a, false, b1, (short)0, acc1, false, false);
        }
        __syncthreads();
    }

    // ---- dump C tiles to LDS: z_lds[row][gate*16 + col] ----
    // C layout: VGPR r, lanes 0-15 -> M=r, lanes 16-31 -> M=r+8; N = lane%16.
    {
        const int mbase = mt * 16 + ((lane >> 4) << 3);
        const int ncol  = lane & 15;
        #pragma unroll
        for (int r = 0; r < 8; ++r) {
            z_lds[(mbase + r) * 64 + (g0 + 0) * 16 + ncol] = acc0[r];
            z_lds[(mbase + r) * 64 + (g0 + 1) * 16 + ncol] = acc1[r];
        }
    }
    __syncthreads();

    // ---- gate math: z order (i, j, f, o); forget bias 1.0 ----
    #pragma unroll
    for (int q = 0; q < 4; ++q) {
        int idx  = q * 256 + tid;        // 0 .. 1023
        int col  = idx & 15;
        int m    = idx >> 4;             // batch row
        int colg = bx * 16 + col;        // global h column

        float iv = z_lds[m * 64 +      col] + bias[           colg];
        float jv = z_lds[m * 64 + 16 + col] + bias[    HDIM + colg];
        float fv = z_lds[m * 64 + 32 + col] + bias[2 * HDIM + colg];
        float ov = z_lds[m * 64 + 48 + col] + bias[3 * HDIM + colg];

        float c_old = c[m * HDIM + colg];
        float c_new = c_old * sigmoidf(fv + 1.0f) + sigmoidf(iv) * tanhf(jv);
        float h_new = tanhf(c_new) * sigmoidf(ov);

        c[m * HDIM + colg]      = c_new;
        h_next[m * HDIM + colg] = h_new;
        out[((size_t)m * TLEN + t) * HDIM + colg] = h_new;
    }
}

// ---------------------------------------------------------------------------
extern "C" void kernel_launch(void* const* d_in, const int* in_sizes, int n_in,
                              void* d_out, int out_size, void* d_ws, size_t ws_size,
                              hipStream_t stream) {
    const float* x    = (const float*)d_in[0];   // [64, 512, 1024]
    const float* W    = (const float*)d_in[1];   // [2048, 4096]
    const float* bias = (const float*)d_in[2];   // [4096]
    float* out = (float*)d_out;                  // [64, 512, 1024]

    // Workspace layout (~16.8 MB):
    //   Wpk : 8388608 bf16 (16 MB)
    //   hbuf: 2 x 65536 f32 ping-pong h slots
    //   cbuf: 65536 f32 cell state
    unsigned short* Wpk = (unsigned short*)d_ws;
    float* hbuf = (float*)((char*)d_ws + (size_t)8388608 * sizeof(unsigned short));
    float* cbuf = hbuf + 2 * (BATCH * HDIM);

    pack_w<<<8388608 / 256, 256, 0, stream>>>(W, Wpk);
    init_state<<<(BATCH * HDIM) / 256, 256, 0, stream>>>(hbuf, cbuf);

    for (int t = 0; t < TLEN; ++t) {
        const float* hp = hbuf + (size_t)(t & 1) * (BATCH * HDIM);
        float*       hn = hbuf + (size_t)((t + 1) & 1) * (BATCH * HDIM);
        lstm_step<<<64, 256, 0, stream>>>(x, Wpk, bias, hp, hn, cbuf, out, t);
    }
}
